// SpatioTemporalBlock_45818711114212
// MI455X (gfx1250) — matrix operand order repaired and checked by reading, not compiled, not run on hardware
//
#include <hip/hip_runtime.h>

typedef _Float16 half16 __attribute__((ext_vector_type(16)));
typedef float   floatx8 __attribute__((ext_vector_type(8)));

#define D_MODEL 512
#define N_HEADS 8
#define D_HEAD  64
#define D_FF    2048
#define MAX_LEN 2000
#define BB      8
#define TT      1024
#define M_ROWS  (BB*TT)   // 8192

// ---------------------------------------------------------------------------
// CDNA5 async global->LDS copy (ASYNCcnt-tracked, no VGPR round trip)
// ---------------------------------------------------------------------------
__device__ __forceinline__ void async_copy_b128(void* lds_dst, const void* gsrc) {
  // generic pointer to a __shared__ object: low 32 bits are the LDS offset
  unsigned l = (unsigned)(unsigned long long)lds_dst;
  asm volatile("global_load_async_to_lds_b128 %0, %1, off"
               :: "v"(l), "v"(gsrc) : "memory");
}
#define WAIT_ASYNC_0() asm volatile("s_wait_asynccnt 0x0" ::: "memory")
#define WAIT_ASYNC_4() asm volatile("s_wait_asynccnt 0x4" ::: "memory")

// ---------------------------------------------------------------------------
// WMMA wrapper: D = A(16x32 f16) * B(32x16 f16) + C(16x16 f32)
// ---------------------------------------------------------------------------
__device__ __forceinline__ floatx8 wmma16(half16 a, half16 b, floatx8 c) {
  return __builtin_amdgcn_wmma_f32_16x16x32_f16(false, a, false, b,
                                                (short)0, c, false, false);
}

// A-fragment (16x32, 16-bit): lane = row m (lane&15); lane-group 0 holds
// K=0..7 & 16..23, group 1 holds 8..15 & 24..31 (ISA 7.12.2).
__device__ __forceinline__ half16 frag_a(const _Float16* base, int ld) {
  const int lane = threadIdx.x & 31;
  const _Float16* p = base + (size_t)(lane & 15) * ld + ((lane >> 4) * 8);
  half16 f;
  ((uint4*)&f)[0] = *(const uint4*)(p);
  ((uint4*)&f)[1] = *(const uint4*)(p + 16);
  return f;
}

// B-fragment (32x16, 16-bit): lane = col n (lane&15); lane-group 0 holds
// K=0..7 & 16..23, group 1 holds 8..15 & 24..31 (same as A).
__device__ __forceinline__ half16 frag_b(const _Float16* base, int ld) {
  const int lane = threadIdx.x & 31;
  const _Float16* p = base + (size_t)(lane & 15) * ld + ((lane >> 4) * 8);
  half16 f;
  ((uint4*)&f)[0] = *(const uint4*)(p);
  ((uint4*)&f)[1] = *(const uint4*)(p + 16);
  return f;
}

// ---------------------------------------------------------------------------
// One-time weight prep: f32 [K][N] -> f16 transposed [N][K] (LDS-tiled)
// ---------------------------------------------------------------------------
__global__ __launch_bounds__(256)
void cvt_transpose_f16(const float* __restrict__ in, _Float16* __restrict__ out,
                       int K, int N) {
  __shared__ float tile[32][33];
  const int n0 = blockIdx.x * 32, k0 = blockIdx.y * 32;
  const int tx = threadIdx.x & 31, ty = threadIdx.x >> 5;  // ty 0..7
#pragma unroll
  for (int i = ty; i < 32; i += 8)
    tile[i][tx] = in[(size_t)(k0 + i) * N + n0 + tx];
  __syncthreads();
#pragma unroll
  for (int i = ty; i < 32; i += 8)
    out[(size_t)(n0 + i) * K + k0 + tx] = (_Float16)tile[tx][i];
}

// ---------------------------------------------------------------------------
// Rel-pos bias collapsed to 1D:  b1d[j] = dot(rpe_table[j+976], rpe_w)
// (bias[q,k] = b1d[q-k+T-1])
// ---------------------------------------------------------------------------
__global__ void rel_bias1d(const float* __restrict__ tab,
                           const float* __restrict__ w,
                           float* __restrict__ out) {
  int j = blockIdx.x * blockDim.x + threadIdx.x;
  if (j >= 2 * TT - 1) return;
  const float* r = tab + (size_t)(j + (MAX_LEN - TT)) * D_HEAD;
  float s = 0.f;
#pragma unroll
  for (int d = 0; d < D_HEAD; ++d) s += r[d] * w[d];
  out[j] = s;
}

// ---------------------------------------------------------------------------
// LayerNorm over D_MODEL=512, one row per block (256 threads), f16 output
// ---------------------------------------------------------------------------
__global__ __launch_bounds__(256)
void layernorm512(const float* __restrict__ x, const float* __restrict__ g,
                  const float* __restrict__ b, _Float16* __restrict__ out) {
  __shared__ float s1[256], s2[256];
  const int row = blockIdx.x, t = threadIdx.x;
  const float* xr = x + (size_t)row * D_MODEL;
  float v0 = xr[t], v1 = xr[t + 256];
  s1[t] = v0 + v1;
  s2[t] = v0 * v0 + v1 * v1;
  __syncthreads();
  for (int s = 128; s > 0; s >>= 1) {
    if (t < s) { s1[t] += s1[t + s]; s2[t] += s2[t + s]; }
    __syncthreads();
  }
  const float mean = s1[0] * (1.f / 512.f);
  const float var  = s2[0] * (1.f / 512.f) - mean * mean;
  const float rstd = rsqrtf(var + 1e-5f);
  _Float16* orow = out + (size_t)row * D_MODEL;
  orow[t]       = (_Float16)((v0 - mean) * rstd * g[t]       + b[t]);
  orow[t + 256] = (_Float16)((v1 - mean) * rstd * g[t + 256] + b[t + 256]);
}

// ---------------------------------------------------------------------------
// Tiled WMMA GEMM: C[M,N] = A16[M,K] * WT16[N,K]^T (+bias, epilogue variants)
//   EPI 0: outh = f16(acc + bias)
//   EPI 1: outh = f16(gelu_exact(acc + bias))
//   EPI 2: outf = res + acc + bias           (f32, fused residual)
// 128x128 block tile, 8 waves (4x2), wave tile 32x64, K-step 32.
// Double-buffered LDS fed by global_load_async_to_lds_b128.
// ---------------------------------------------------------------------------
template <int EPI>
__global__ __launch_bounds__(256)
void gemm_wmma_f16(const _Float16* __restrict__ A, const _Float16* __restrict__ WT,
                   const float* __restrict__ bias, const float* __restrict__ res,
                   float* __restrict__ outf, _Float16* __restrict__ outh,
                   int M, int N, int K) {
  __shared__ _Float16 As[2][128][40];  // A tile, row-major (padded rows)
  __shared__ _Float16 Wt[2][128][40];  // W tile, [n][k] (WT is pre-transposed)
  const int t = threadIdx.x, wave = t >> 5, lane = t & 31;
  const int wm = wave >> 1, wn = wave & 1;
  const int row0 = blockIdx.y * 128, col0 = blockIdx.x * 128;
  floatx8 acc[2][4] = {};
  const int nk = K / 32;

  // staging: thread t copies 32B of A row (r = t>>1) and 32B of WT row r.
  const int sr = t >> 1, ss = (t & 1) * 16;
  auto stage = [&](int kt, int buf) {
    const _Float16* ga = A + (size_t)(row0 + sr) * K + kt * 32 + ss;
    async_copy_b128(&As[buf][sr][ss],     ga);
    async_copy_b128(&As[buf][sr][ss + 8], ga + 8);
    const _Float16* gw = WT + (size_t)(col0 + sr) * K + kt * 32 + ss;
    async_copy_b128(&Wt[buf][sr][ss],     gw);
    async_copy_b128(&Wt[buf][sr][ss + 8], gw + 8);
  };

  stage(0, 0);
  for (int kt = 0; kt < nk; ++kt) {
    const int cur = kt & 1;
    if (kt + 1 < nk) {  // prefetch next tile into the other buffer
      stage(kt + 1, cur ^ 1);
      WAIT_ASYNC_4();   // tile kt complete (in-order), tile kt+1 in flight
    } else {
      WAIT_ASYNC_0();
    }
    __syncthreads();

    half16 af[2];
#pragma unroll
    for (int mi = 0; mi < 2; ++mi)
      af[mi] = frag_a(&As[cur][wm * 32 + mi * 16][0], 40);
#pragma unroll
    for (int ni = 0; ni < 4; ++ni) {
      half16 bf = frag_b(&Wt[cur][wn * 64 + ni * 16][0], 40);
#pragma unroll
      for (int mi = 0; mi < 2; ++mi) acc[mi][ni] = wmma16(af[mi], bf, acc[mi][ni]);
    }
    __syncthreads();  // done reading buf[cur] before it is refilled at kt+2
  }

  // Epilogue: C layout -> n = lane&15, m = (lane>>4)*8 + v
#pragma unroll
  for (int mi = 0; mi < 2; ++mi)
#pragma unroll
    for (int ni = 0; ni < 4; ++ni)
#pragma unroll
      for (int v = 0; v < 8; ++v) {
        const int r = row0 + wm * 32 + mi * 16 + ((lane >> 4) * 8) + v;
        const int c = col0 + wn * 64 + ni * 16 + (lane & 15);
        float val = acc[mi][ni][v] + bias[c];
        if constexpr (EPI == 1)
          val = 0.5f * val * (1.f + erff(val * 0.70710678118654752f));
        if constexpr (EPI == 2)
          outf[(size_t)r * N + c] = res[(size_t)r * N + c] + val;
        else
          outh[(size_t)r * N + c] = (_Float16)val;
      }
}

// ---------------------------------------------------------------------------
// Flash attention (per b,h): 128 q-rows/block, 8 waves x 16 rows,
// streaming 64-key K/V tiles.  S = Q K^T and O += P V via WMMA.
// ---------------------------------------------------------------------------
template <bool HAS_BIAS>
__global__ __launch_bounds__(256)
void attn_wmma(const _Float16* __restrict__ qkv, const float* __restrict__ bias1d,
               const unsigned char* __restrict__ mask, _Float16* __restrict__ outp) {
  __shared__ _Float16 Kls[64][64];     // K tile row-major: [key][d]
  __shared__ _Float16 Vt[64][64];      // V tile transposed: [d][key]
  __shared__ _Float16 Pls[8][16][64];  // per-wave P strip: [wave][m][key]
  __shared__ float    bsh[2048];       // rel-bias 1D vector
  __shared__ float    msk[64];         // mask addend per key tile
  const int t = threadIdx.x, wave = t >> 5, lane = t & 31;
  const int qt = blockIdx.x, h = blockIdx.y, b = blockIdx.z;
  const int nlane = lane & 15;

  if (HAS_BIAS)
    for (int i = t; i < 2 * TT - 1; i += 256) bsh[i] = bias1d[i];

  // Q fragments for this wave's 16 rows (loaded once, straight from global)
  const size_t qrow0 = (size_t)b * TT + qt * 128 + wave * 16;
  half16 aq[2];
#pragma unroll
  for (int c = 0; c < 2; ++c)
    aq[c] = frag_a(qkv + qrow0 * (3 * D_MODEL) + h * D_HEAD + c * 32, 3 * D_MODEL);

  floatx8 O[4] = {};
  float mrow[8], lrow[8];
#pragma unroll
  for (int v = 0; v < 8; ++v) { mrow[v] = -1e30f; lrow[v] = 0.f; }
  const int qg_base = qt * 128 + wave * 16 + ((lane >> 4) * 8);  // +v -> global q

  for (int j = 0; j < TT / 64; ++j) {
    __syncthreads();
    {  // stage K async (row-major), V via VGPRs (needs transpose), mask
      const int r = t >> 2, d0 = (t & 3) * 16;
      const size_t kvrow = (size_t)b * TT + j * 64 + r;
      const _Float16* ks = qkv + kvrow * (3 * D_MODEL) + D_MODEL + h * D_HEAD + d0;
      async_copy_b128(&Kls[r][d0],     ks);
      async_copy_b128(&Kls[r][d0 + 8], ks + 8);
      const _Float16* vs = qkv + kvrow * (3 * D_MODEL) + 2 * D_MODEL + h * D_HEAD + d0;
      _Float16 tmp[16];
      ((uint4*)tmp)[0] = ((const uint4*)vs)[0];
      ((uint4*)tmp)[1] = ((const uint4*)vs)[1];
#pragma unroll
      for (int i = 0; i < 16; ++i) Vt[d0 + i][r] = tmp[i];
      if (HAS_BIAS && t < 64)
        msk[t] = mask[(size_t)b * TT + j * 64 + t] ? 0.f : -1e30f;
    }
    WAIT_ASYNC_0();
    __syncthreads();

    // S strip: 16 q x 64 k (4 n-tiles, 2 k-steps over d)
    floatx8 S[4] = {};
#pragma unroll
    for (int jj = 0; jj < 4; ++jj)
#pragma unroll
      for (int c = 0; c < 2; ++c) {
        half16 bk = frag_b(&Kls[jj * 16][c * 32], 64);
        S[jj] = wmma16(aq[c], bk, S[jj]);
      }

    // scale + rel-bias + mask
#pragma unroll
    for (int jj = 0; jj < 4; ++jj) {
      const int kg = j * 64 + jj * 16 + nlane;
#pragma unroll
      for (int v = 0; v < 8; ++v) {
        float s = S[jj][v] * 0.125f;  // 1/sqrt(64)
        if (HAS_BIAS) {
          s += bsh[(qg_base + v) - kg + (TT - 1)];
          s += msk[jj * 16 + nlane];
        }
        S[jj][v] = s;
      }
    }

    // online softmax (row spread across 16 lanes of a half-wave)
#pragma unroll
    for (int v = 0; v < 8; ++v) {
      float mx = fmaxf(fmaxf(S[0][v], S[1][v]), fmaxf(S[2][v], S[3][v]));
#pragma unroll
      for (int d = 1; d < 16; d <<= 1) mx = fmaxf(mx, __shfl_xor(mx, d));
      const float mnew  = fmaxf(mrow[v], mx);
      const float alpha = __expf(mrow[v] - mnew);
      float ps = 0.f;
#pragma unroll
      for (int jj = 0; jj < 4; ++jj) {
        float p = __expf(S[jj][v] - mnew);
        S[jj][v] = p;
        ps += p;
      }
#pragma unroll
      for (int d = 1; d < 16; d <<= 1) ps += __shfl_xor(ps, d);
      lrow[v] = lrow[v] * alpha + ps;
      mrow[v] = mnew;
#pragma unroll
      for (int dd = 0; dd < 4; ++dd) O[dd][v] *= alpha;
    }

    // P (C-layout) -> wave-private LDS strip, then re-read as A fragments
#pragma unroll
    for (int jj = 0; jj < 4; ++jj)
#pragma unroll
      for (int v = 0; v < 8; ++v)
        Pls[wave][(lane >> 4) * 8 + v][jj * 16 + nlane] = (_Float16)S[jj][v];
    asm volatile("s_wait_dscnt 0x0" ::: "memory");  // wave-local LDS RAW fence

#pragma unroll
    for (int c = 0; c < 2; ++c) {
      half16 pa = frag_a(&Pls[wave][0][c * 32], 64);
#pragma unroll
      for (int dd = 0; dd < 4; ++dd) {
        half16 bv = frag_b(&Vt[dd * 16][c * 32], 64);
        O[dd] = wmma16(pa, bv, O[dd]);
      }
    }
  }

  // epilogue: O / l -> f16 output [B*T, D_MODEL] at head offset
#pragma unroll
  for (int dd = 0; dd < 4; ++dd)
#pragma unroll
    for (int v = 0; v < 8; ++v) {
      const size_t r = qrow0 + (lane >> 4) * 8 + v;
      const int c = h * D_HEAD + dd * 16 + nlane;
      outp[r * D_MODEL + c] = (_Float16)(O[dd][v] / lrow[v]);
    }
}

// ---------------------------------------------------------------------------
// Launch
// ---------------------------------------------------------------------------
extern "C" void kernel_launch(void* const* d_in, const int* in_sizes, int n_in,
                              void* d_out, int out_size, void* d_ws, size_t ws_size,
                              hipStream_t stream) {
  (void)in_sizes; (void)n_in; (void)out_size; (void)ws_size;
  const float* x        = (const float*)d_in[0];
  const unsigned char* mask = (const unsigned char*)d_in[1];
  const float* ns_g = (const float*)d_in[2],  *ns_b = (const float*)d_in[3];
  const float* nt_g = (const float*)d_in[4],  *nt_b = (const float*)d_in[5];
  const float* nf_g = (const float*)d_in[6],  *nf_b = (const float*)d_in[7];
  const float* s_qkv_w = (const float*)d_in[8],  *s_qkv_b = (const float*)d_in[9];
  const float* s_proj_w = (const float*)d_in[10], *s_proj_b = (const float*)d_in[11];
  const float* t_qkv_w = (const float*)d_in[12], *t_qkv_b = (const float*)d_in[13];
  const float* t_proj_w = (const float*)d_in[14], *t_proj_b = (const float*)d_in[15];
  const float* rpe_table = (const float*)d_in[16], *rpe_w = (const float*)d_in[17];
  const float* ff_w1 = (const float*)d_in[18], *ff_b1 = (const float*)d_in[19];
  const float* ff_w2 = (const float*)d_in[20], *ff_b2 = (const float*)d_in[21];

  char* base = (char*)d_ws;
  size_t off = 0;
  auto alloc = [&](size_t bytes) -> void* {
    void* p = base + off;
    off = (off + bytes + 255) & ~(size_t)255;
    return p;
  };
  // all weights stored transposed [N][K] in f16
  _Float16* wqs   = (_Float16*)alloc((size_t)1536 * 512 * 2);
  _Float16* wps   = (_Float16*)alloc((size_t)512 * 512 * 2);
  _Float16* wqt   = (_Float16*)alloc((size_t)1536 * 512 * 2);
  _Float16* wpt   = (_Float16*)alloc((size_t)512 * 512 * 2);
  _Float16* w1h   = (_Float16*)alloc((size_t)2048 * 512 * 2);
  _Float16* w2h   = (_Float16*)alloc((size_t)512 * 2048 * 2);
  _Float16* ln16  = (_Float16*)alloc((size_t)M_ROWS * 512 * 2);
  _Float16* qkv16 = (_Float16*)alloc((size_t)M_ROWS * 1536 * 2);
  _Float16* attn16= (_Float16*)alloc((size_t)M_ROWS * 512 * 2);
  _Float16* h16   = (_Float16*)alloc((size_t)M_ROWS * 2048 * 2);
  float* x1  = (float*)alloc((size_t)M_ROWS * 512 * 4);
  float* x2  = (float*)alloc((size_t)M_ROWS * 512 * 4);
  float* b1d = (float*)alloc(2048 * 4);
  float* outf = (float*)d_out;

  auto cvtT = [&](const float* s, _Float16* d, int K, int N) {
    cvt_transpose_f16<<<dim3(N / 32, K / 32), 256, 0, stream>>>(s, d, K, N);
  };
  cvtT(s_qkv_w, wqs, 512, 1536);
  cvtT(s_proj_w, wps, 512, 512);
  cvtT(t_qkv_w, wqt, 512, 1536);
  cvtT(t_proj_w, wpt, 512, 512);
  cvtT(ff_w1, w1h, 512, 2048);
  cvtT(ff_w2, w2h, 2048, 512);
  rel_bias1d<<<8, 256, 0, stream>>>(rpe_table, rpe_w, b1d);

  // ---- spatial attention sub-block ----
  layernorm512<<<M_ROWS, 256, 0, stream>>>(x, ns_g, ns_b, ln16);
  gemm_wmma_f16<0><<<dim3(1536 / 128, M_ROWS / 128), 256, 0, stream>>>(
      ln16, wqs, s_qkv_b, nullptr, nullptr, qkv16, M_ROWS, 1536, 512);
  attn_wmma<false><<<dim3(TT / 128, N_HEADS, BB), 256, 0, stream>>>(
      qkv16, nullptr, nullptr, attn16);
  gemm_wmma_f16<2><<<dim3(512 / 128, M_ROWS / 128), 256, 0, stream>>>(
      attn16, wps, s_proj_b, x, x1, nullptr, M_ROWS, 512, 512);

  // ---- temporal attention sub-block (rel-bias + mask) ----
  layernorm512<<<M_ROWS, 256, 0, stream>>>(x1, nt_g, nt_b, ln16);
  gemm_wmma_f16<0><<<dim3(1536 / 128, M_ROWS / 128), 256, 0, stream>>>(
      ln16, wqt, t_qkv_b, nullptr, nullptr, qkv16, M_ROWS, 1536, 512);
  attn_wmma<true><<<dim3(TT / 128, N_HEADS, BB), 256, 0, stream>>>(
      qkv16, b1d, mask, attn16);
  gemm_wmma_f16<2><<<dim3(512 / 128, M_ROWS / 128), 256, 0, stream>>>(
      attn16, wpt, t_proj_b, x1, x2, nullptr, M_ROWS, 512, 512);

  // ---- FFN sub-block (exact GELU fused into GEMM1 epilogue) ----
  layernorm512<<<M_ROWS, 256, 0, stream>>>(x2, nf_g, nf_b, ln16);
  gemm_wmma_f16<1><<<dim3(2048 / 128, M_ROWS / 128), 256, 0, stream>>>(
      ln16, w1h, ff_b1, nullptr, nullptr, h16, M_ROWS, 2048, 512);
  gemm_wmma_f16<2><<<dim3(512 / 128, M_ROWS / 128), 256, 0, stream>>>(
      h16, w2h, ff_b2, x2, outf, nullptr, M_ROWS, 512, 2048);
}
